// FeedForwardQuantum_65481071395866
// MI455X (gfx1250) — matrix operand order repaired and checked by reading, not compiled, not run on hardware
//
#include <hip/hip_runtime.h>

// ---------------------------------------------------------------------------
// FeedForwardQuantum, fused for MI455X (gfx1250, wave32, WMMA bf16 + TDM).
//
// Closed form of the quantum layer:
//   m_j  = cos(theta_j) * cos(x_j)
//   q[k] = prod_{j<=k} m_j  (k>=1),   q[0] = prod_{j>=1} m_j
// Then h = relu(q @ w1^T + b1) kept in LDS as bf16, out = h @ w2^T + b2.
// GEMM2 B-panels (w2^T, 32x256 bf16 per K-step) are DMA'd into LDS by the
// Tensor Data Mover with double buffering, synced via TENSORcnt.
// ---------------------------------------------------------------------------

typedef __bf16 bf16;
typedef bf16     v16bf __attribute__((ext_vector_type(16)));
typedef bf16     v8bf  __attribute__((ext_vector_type(8)));
typedef float    v8f   __attribute__((ext_vector_type(8)));
typedef unsigned u32x4 __attribute__((ext_vector_type(4)));
typedef int      i32x4 __attribute__((ext_vector_type(4)));
typedef int      i32x8 __attribute__((ext_vector_type(8)));

#define E_DIM     256
#define NQ_DIM    10
#define FF_DIM    1024
#define TOK_TILE  128              // tokens per block
#define HS_STRIDE 1032             // bf16/row of H (1024 + 8 pad): conflict-free b128
#define QS_STRIDE 40               // bf16/row of q (32 used + pad)
#define BS_STRIDE 264              // bf16/row of B panel (256 + TDM pad of 4 dwords)

#define HS_BYTES  (TOK_TILE * HS_STRIDE * 2)        // 264192
#define QS_BYTES  (TOK_TILE * QS_STRIDE * 2)        // 10240
#define BS_BYTES  (32 * BS_STRIDE * 2)              // 16896 per buffer
#define BS_OFF    (HS_BYTES + QS_BYTES)             // LDS byte offset of panel 0
#define SMEM_BYTES (BS_OFF + 2 * BS_BYTES)          // 308224 B < 320 KB WGP LDS

union V16 { v8bf h[2]; v16bf v; };

static __device__ inline v8f vzero8() {
  v8f t;
#pragma unroll
  for (int i = 0; i < 8; ++i) t[i] = 0.f;
  return t;
}

// TDM: DMA one 32x256 bf16 panel of w2t (rows k0..k0+31) into LDS at ldsOff.
// D# per CDNA5 ISA ch.8: 2D tile, data_size=2B, LDS pad 4 dwords every 128
// dwords -> landed row stride 528 B (bank-conflict-free for the B-frag reads).
static __device__ inline void tdm_load_panel(const bf16* gsrc, unsigned ldsOff) {
  unsigned long long ga = (unsigned long long)(size_t)gsrc;
  u32x4 g0;
  g0[0] = 1u;                                           // count=1, no gather
  g0[1] = ldsOff;                                       // lds_addr
  g0[2] = (unsigned)(ga & 0xFFFFFFFFu);                 // global_addr[31:0]
  g0[3] = (unsigned)((ga >> 32) & 0x01FFFFFFu)          // global_addr[56:32]
        | (2u << 30);                                   // type = 2 ("image")
  i32x8 g1;
  g1[0] = (1 << 16)                                     // data_size = 2 bytes
        | (1 << 20)                                     // pad_enable
        | (6 << 22)                                     // pad_interval: 128 dwords
        | (3 << 25);                                    // pad_amount:   4 dwords
  g1[1] = (int)(256u << 16);                            // tensor_dim0 = 256
  g1[2] = (int)(1024u << 16);                           // tensor_dim1 = 1024
  g1[3] = (int)(256u << 16);                            // tile_dim0 = 256
  g1[4] = 32;                                           // tile_dim1 = 32
  g1[5] = 256;                                          // tensor_dim0_stride = 256
  g1[6] = 0;
  g1[7] = 0;
  i32x4 z4 = {0, 0, 0, 0};
#if __has_include(<hip/amd_detail/amd_gfx1250_TDM.h>)
  i32x8 z8 = {0, 0, 0, 0, 0, 0, 0, 0};
  __builtin_amdgcn_tensor_load_to_lds(g0, g1, z4, z4, z8, 0);   // 6-arg toolchain
#else
  __builtin_amdgcn_tensor_load_to_lds(g0, g1, z4, z4, 0);       // ROCm 7.2 (5-arg)
#endif
}

// Prep: w1t[k][f] = bf16(w1[f][k]) with K padded 10->32; w2t[k][n] = bf16(w2[n][k]).
__global__ void __launch_bounds__(256) qffn_prep_kernel(
    const float* __restrict__ w1, const float* __restrict__ w2,
    bf16* __restrict__ w1t, bf16* __restrict__ w2t) {
  int idx = blockIdx.x * 256 + threadIdx.x;
  const int W2T_N = FF_DIM * E_DIM;   // 262144
  const int W1T_N = 32 * FF_DIM;      // 32768
  if (idx < W2T_N) {
    int k = idx >> 8;                 // 0..1023
    int n = idx & 255;                // 0..255
    w2t[idx] = (bf16)w2[n * FF_DIM + k];
  } else if (idx < W2T_N + W1T_N) {
    int i = idx - W2T_N;
    int k = i >> 10;                  // 0..31
    int f = i & 1023;                 // 0..1023
    float v = (k < NQ_DIM) ? w1[f * NQ_DIM + k] : 0.f;
    w1t[i] = (bf16)v;
  }
}

__global__ void __launch_bounds__(256) qffn_fused_kernel(
    const float* __restrict__ x, const float* __restrict__ theta,
    const float* __restrict__ b1, const float* __restrict__ b2,
    const bf16* __restrict__ w1t, const bf16* __restrict__ w2t,
    float* __restrict__ out) {
  extern __shared__ char smem[];
  bf16* Hs  = (bf16*)smem;                  // [128][HS_STRIDE] bf16
  bf16* qsb = (bf16*)(smem + HS_BYTES);     // [128][QS_STRIDE] bf16, cols 10..31 = 0
  bf16* Bs  = (bf16*)(smem + BS_OFF);       // [2][32][BS_STRIDE] bf16 (TDM target)

  const int tid  = threadIdx.x;
  const int lane = tid & 31;
  const int wid  = tid >> 5;                // 8 waves
  const int g    = lane >> 4;               // half-wave group
  const int ln   = lane & 15;
  const size_t tokBase = (size_t)blockIdx.x * TOK_TILE;

  // Kick off the DMA of the first w2t K-panel; it overlaps stages A and B.
  if (wid == 0) tdm_load_panel(w2t, BS_OFF);

  // ---- Stage A: analytic quantum layer -> q (bf16, K-padded to 32) ----
  if (tid < TOK_TILE) {
    const float* xr = x + (tokBase + tid) * E_DIM;
    float m[NQ_DIM];
#pragma unroll
    for (int j = 0; j < NQ_DIM; ++j)
      m[j] = __cosf(theta[j]) * __cosf(xr[j]);
    bf16* qrow = qsb + tid * QS_STRIDE;
    float pref = 1.f, suf = 1.f;
#pragma unroll
    for (int j = 0; j < NQ_DIM; ++j) {
      pref *= m[j];
      if (j > 0) { qrow[j] = (bf16)pref; suf *= m[j]; }
    }
    qrow[0] = (bf16)suf;
#pragma unroll
    for (int j = NQ_DIM; j < 32; ++j) qrow[j] = (bf16)0.f;
  }
  __syncthreads();

  // ---- Stage B: H = relu(q @ w1^T + b1), WMMA, one M-tile per wave ----
  {
    const int mt = wid;                     // 8 waves == 8 M tiles of 16 tokens
    const bf16* arow = qsb + (mt * 16 + ln) * QS_STRIDE;
    V16 ua;                                 // A 16x32 bf16 frag (ISA 7.12.2 layout)
    ua.h[0] = *(const v8bf*)(arow + g * 8);
    ua.h[1] = *(const v8bf*)(arow + 16 + g * 8);
    const v16bf A = ua.v;
#pragma unroll 4
    for (int nt = 0; nt < FF_DIM / 16; ++nt) {
      const bf16* bp = w1t + nt * 16 + ln;   // lane = N, elements = K (table order)
      V16 ub;
#pragma unroll
      for (int s = 0; s < 8; ++s) {
        ub.v[s]     = bp[(g * 8 + s) * FF_DIM];
        ub.v[8 + s] = bp[(16 + g * 8 + s) * FF_DIM];
      }
      v8f c = vzero8();
      c = __builtin_amdgcn_wmma_f32_16x16x32_bf16(false, A, false, ub.v,
                                                  (short)0, c, false, false);
      const float bias = b1[nt * 16 + ln];
      bf16* hbase = Hs + nt * 16 + ln;
#pragma unroll
      for (int r = 0; r < 8; ++r) {
        float hv = c[r] + bias;
        hv = hv > 0.f ? hv : 0.f;
        hbase[(mt * 16 + g * 8 + r) * HS_STRIDE] = (bf16)hv;
      }
    }
  }
  __syncthreads();

  // ---- Stage C: out = H @ w2^T + b2, WMMA over K=1024, TDM double buffer ----
  {
    const int nt0 = wid * 2, nt1 = nt0 + 1;   // each wave: 2 N tiles x all 8 M tiles
    v8f acc0[8], acc1[8];
#pragma unroll
    for (int mtI = 0; mtI < 8; ++mtI) { acc0[mtI] = vzero8(); acc1[mtI] = vzero8(); }

    int buf = 0;
#pragma unroll 1
    for (int k0 = 0; k0 < FF_DIM; k0 += 32, buf ^= 1) {
      if (wid == 0) {
        if (k0 + 32 < FF_DIM) {
          // DMA next panel into the other buffer, then ensure current landed.
          tdm_load_panel(w2t + (size_t)(k0 + 32) * E_DIM,
                         BS_OFF + (unsigned)((buf ^ 1) * BS_BYTES));
          __builtin_amdgcn_s_wait_tensorcnt((short)1);
        } else {
          __builtin_amdgcn_s_wait_tensorcnt((short)0);
        }
      }
      __syncthreads();                       // publish current panel to all waves

      const bf16* bp0 = Bs + buf * (32 * BS_STRIDE) + nt0 * 16 + ln;
      V16 ub0, ub1;                          // B frags: lane = N, elements = K (table order)
#pragma unroll
      for (int s = 0; s < 8; ++s) {
        ub0.v[s]     = bp0[(g * 8 + s) * BS_STRIDE];
        ub0.v[8 + s] = bp0[(16 + g * 8 + s) * BS_STRIDE];
        ub1.v[s]     = bp0[(g * 8 + s) * BS_STRIDE + 16];
        ub1.v[8 + s] = bp0[(16 + g * 8 + s) * BS_STRIDE + 16];
      }
#pragma unroll
      for (int mtI = 0; mtI < 8; ++mtI) {
        const bf16* arow = Hs + (mtI * 16 + ln) * HS_STRIDE + k0;
        V16 ua;
        ua.h[0] = *(const v8bf*)(arow + g * 8);
        ua.h[1] = *(const v8bf*)(arow + 16 + g * 8);
        acc0[mtI] = __builtin_amdgcn_wmma_f32_16x16x32_bf16(
            false, ua.v, false, ub0.v, (short)0, acc0[mtI], false, false);
        acc1[mtI] = __builtin_amdgcn_wmma_f32_16x16x32_bf16(
            false, ua.v, false, ub1.v, (short)0, acc1[mtI], false, false);
      }
      __syncthreads();                       // reads done before next DMA overwrite
    }

    const float bias0 = b2[nt0 * 16 + ln];
    const float bias1 = b2[nt1 * 16 + ln];
#pragma unroll
    for (int mtI = 0; mtI < 8; ++mtI) {
#pragma unroll
      for (int r = 0; r < 8; ++r) {
        const size_t row = tokBase + mtI * 16 + g * 8 + r;   // D layout: M = 8*g + r
        out[row * E_DIM + nt0 * 16 + ln] = acc0[mtI][r] + bias0;
        out[row * E_DIM + nt1 * 16 + ln] = acc1[mtI][r] + bias1;
      }
    }
  }
}

extern "C" void kernel_launch(void* const* d_in, const int* in_sizes, int n_in,
                              void* d_out, int out_size, void* d_ws, size_t ws_size,
                              hipStream_t stream) {
  const float* x     = (const float*)d_in[0];
  const float* theta = (const float*)d_in[1];
  const float* w1    = (const float*)d_in[2];
  const float* b1    = (const float*)d_in[3];
  const float* w2    = (const float*)d_in[4];
  const float* b2    = (const float*)d_in[5];
  float* out = (float*)d_out;

  bf16* w2t = (bf16*)d_ws;                                        // 512 KB
  bf16* w1t = (bf16*)((char*)d_ws + (size_t)FF_DIM * E_DIM * 2);  // 64 KB

  const int prepN = FF_DIM * E_DIM + 32 * FF_DIM;                 // 294912 elems
  qffn_prep_kernel<<<(prepN + 255) / 256, 256, 0, stream>>>(w1, w2, w1t, w2t);

  const int nBlocks = (8 * 2048) / TOK_TILE;                      // 128 blocks
  qffn_fused_kernel<<<nBlocks, 256, SMEM_BYTES, stream>>>(
      x, theta, b1, b2, w1t, w2t, out);
}
